// KBLaMBitNetDecoderLayer_52982716563903
// MI455X (gfx1250) — compile-verified, matrix operand order checked by reading, not run on hardware
//
#include <hip/hip_runtime.h>
#include <hip/hip_bf16.h>
#include <cstdint>
#include <cstddef>

// ---------------------------------------------------------------------------
// Problem dimensions (fixed by the reference)
// ---------------------------------------------------------------------------
#define S_LEN   2048
#define H_DIM   2048
#define NH      32
#define NKV     8
#define GROUPS  4          // NH / NKV
#define HD      64
#define DFF     8192
#define NKB     256
#define KB_LEN  257        // sink + NKB
#define KB_PAD  288        // KB_LEN rounded up to 32, zero padded
#define S_PAD   2080       // S_LEN + 32 pad for key-chunk overreach
#define LDB     72         // padded LDS leading dim (breaks 128B bank stride)

typedef __bf16 bf16;
typedef __attribute__((ext_vector_type(8)))  __bf16 v8bf;
typedef __attribute__((ext_vector_type(16))) __bf16 v16bf;
typedef __attribute__((ext_vector_type(8)))  float  v8f;

union FragBF { v16bf v; v8bf h[2]; };

// ---------------------------------------------------------------------------
// WMMA helpers.
// A fragment (16xK tile, row-major, K contiguous): lanes 0-15 hold rows 0..15
// with K groups {k0..k0+7, k0+16..k0+23}; lanes 16-31 hold the same rows with
// K groups {k0+8..k0+15, k0+24..k0+31}.  B fragment is identical with
// row -> column (the "N x K" matrix, i.e. B^T row-major).
// ---------------------------------------------------------------------------
__device__ __forceinline__ v16bf load_frag(const bf16* __restrict__ base, int ld,
                                           int rc0, int k0, int lane) {
    const int idx = rc0 + (lane & 15);
    const int hg  = (lane >> 4) * 8;
    const bf16* p = base + (size_t)idx * (size_t)ld + (size_t)(k0 + hg);
    FragBF f;
    f.h[0] = *reinterpret_cast<const v8bf*>(p);
    f.h[1] = *reinterpret_cast<const v8bf*>(p + 16);
    return f.v;
}

__device__ __forceinline__ v8f wmma_bf16(v16bf a, v16bf b, v8f c) {
    return __builtin_amdgcn_wmma_f32_16x16x32_bf16(false, a, false, b,
                                                   (short)0, c, false, false);
}

// Per-lane async 16B copy global -> LDS (tracked by ASYNCcnt).
__device__ __forceinline__ void async_copy16(const bf16* g, uint32_t lds_addr) {
    const unsigned long long ga = (unsigned long long)(uintptr_t)g;
    asm volatile("global_load_async_to_lds_b128 %0, %1, off"
                 :: "v"(lds_addr), "v"(ga) : "memory");
}

__device__ __forceinline__ void wait_async0() {
    asm volatile("s_wait_asynccnt 0" ::: "memory");
}

// ---------------------------------------------------------------------------
// Generic bf16 GEMM:  C[M,N] = A[M,K] * B[N,K]^T  (+ residual)
// Block = 256 threads = 8 waves; block tile 128x64, K-step 64.
// The shared 64(col) x 64(K) B tile is staged once per block into LDS via
// async copies (double buffered); each wave reads B fragments from LDS and
// streams its private A fragments from global.
// M and N are multiples of 128 / 64 for every call site.
// ---------------------------------------------------------------------------
__device__ __forceinline__ void stage_b_tile(const bf16* __restrict__ B, int K,
                                             int n0, int k0, uint32_t lds_base,
                                             int tid)
{
#pragma unroll
    for (int r = 0; r < 2; ++r) {
        const int c    = tid + r * 256;     // 0..511 : 16B chunks
        const int col  = c >> 3;            // 0..63
        const int koff = (c & 7) * 8;       // 0..56
        const bf16* g = B + (size_t)(n0 + col) * (size_t)K + (size_t)(k0 + koff);
        async_copy16(g, lds_base + (uint32_t)(col * LDB + koff) * 2u);
    }
}

__global__ __launch_bounds__(256) void gemm_bf16_nt(
    const bf16* __restrict__ A, const bf16* __restrict__ B,
    float* __restrict__ C, int M, int N, int K,
    const float* __restrict__ residual)
{
    __shared__ __align__(16) bf16 bsh[2][64 * LDB];
    const int tid  = threadIdx.x;
    const int lane = tid & 31;
    const int wave = tid >> 5;
    const int m0 = blockIdx.x * 128 + wave * 16;
    const int n0 = blockIdx.y * 64;

    const uint32_t lds0 = (uint32_t)(uintptr_t)(&bsh[0][0]);
    const uint32_t lds1 = (uint32_t)(uintptr_t)(&bsh[1][0]);

    stage_b_tile(B, K, n0, 0, lds0, tid);

    v8f acc0 = {}, acc1 = {}, acc2 = {}, acc3 = {};
    int buf = 0;
    for (int k0 = 0; k0 < K; k0 += 64) {
        wait_async0();
        __syncthreads();                    // tile[buf] visible to all waves
        if (k0 + 64 < K)
            stage_b_tile(B, K, n0, k0 + 64, buf ? lds0 : lds1, tid);
        const bf16* bt = &bsh[buf][0];
#pragma unroll
        for (int kk = 0; kk < 64; kk += 32) {
            // materialize all fragments first so the 8 ds_load_b128 issue as a
            // batch (single dscnt wait) instead of serializing against WMMAs
            v16bf a  = load_frag(A, K, m0, k0 + kk, lane);
            v16bf b0 = load_frag(bt, LDB,  0, kk, lane);
            v16bf b1 = load_frag(bt, LDB, 16, kk, lane);
            v16bf b2 = load_frag(bt, LDB, 32, kk, lane);
            v16bf b3 = load_frag(bt, LDB, 48, kk, lane);
            acc0 = wmma_bf16(a, b0, acc0);
            acc1 = wmma_bf16(a, b1, acc1);
            acc2 = wmma_bf16(a, b2, acc2);
            acc3 = wmma_bf16(a, b3, acc3);
        }
        buf ^= 1;
    }

    const int cl = lane & 15;
    const int hg = (lane >> 4) * 8;
    v8f accs[4] = {acc0, acc1, acc2, acc3};
#pragma unroll
    for (int t = 0; t < 4; ++t) {
#pragma unroll
        for (int i = 0; i < 8; ++i) {
            const int row = m0 + hg + i;
            const int col = n0 + 16 * t + cl;
            float v = accs[t][i];
            if (residual) v += residual[(size_t)row * N + col];
            C[(size_t)row * N + col] = v;
        }
    }
}

// ---------------------------------------------------------------------------
// Elementwise kernels
// ---------------------------------------------------------------------------
__global__ void cvt_bf16_kernel(const float* __restrict__ src,
                                bf16* __restrict__ dst, size_t n) {
    size_t i = (size_t)blockIdx.x * blockDim.x + threadIdx.x;
    if (i < n) dst[i] = (bf16)src[i];
}

__global__ void sqrelu_bf16_kernel(const float* __restrict__ src,
                                   bf16* __restrict__ dst, size_t n) {
    size_t i = (size_t)blockIdx.x * blockDim.x + threadIdx.x;
    if (i < n) {
        float v = src[i];
        v = v > 0.0f ? v * v : 0.0f;
        dst[i] = (bf16)v;
    }
}

// RMSNorm over H, output bf16. One block (256 thr) per row.
__global__ __launch_bounds__(256) void rmsnorm_kernel(
    const float* __restrict__ x, const float* __restrict__ w,
    bf16* __restrict__ out)
{
    __shared__ float red[256];
    const int row = blockIdx.x;
    const float* xr = x + (size_t)row * H_DIM;
    float s = 0.0f;
    for (int i = threadIdx.x; i < H_DIM; i += 256) {
        float v = xr[i];
        s += v * v;
    }
    red[threadIdx.x] = s;
    __syncthreads();
    for (int off = 128; off > 0; off >>= 1) {
        if ((int)threadIdx.x < off) red[threadIdx.x] += red[threadIdx.x + off];
        __syncthreads();
    }
    const float inv = rsqrtf(red[0] / (float)H_DIM + 1e-5f);
    for (int i = threadIdx.x; i < H_DIM; i += 256)
        out[(size_t)row * H_DIM + i] = (bf16)(xr[i] * inv * w[i]);
}

// [S, nheads*64] fp32 -> [nheads][s_pad][64] bf16, optional RoPE, zero pad.
__global__ void rope_transpose_kernel(const float* __restrict__ src,
                                      bf16* __restrict__ dst,
                                      const int* __restrict__ pos_ids,
                                      int nheads, int s_pad, int do_rope)
{
    const int s = blockIdx.x;
    const int h = blockIdx.y;
    const int d = threadIdx.x;           // 0..63
    bf16* out = dst + ((size_t)h * s_pad + s) * HD + d;
    if (s >= S_LEN) { *out = (bf16)0.0f; return; }
    const size_t base = (size_t)s * nheads * HD + (size_t)h * HD;
    float v = src[base + d];
    if (do_rope) {
        const float pos = (float)pos_ids[s];
        const int dm = d & 31;
        // theta^(-2*dm/64) = exp(-2*dm/64 * ln(10000))
        const float inv_freq = __expf(-(float)(2 * dm) * (9.210340372f / 64.0f));
        const float ang = pos * inv_freq;
        const float c = __cosf(ang), sn = __sinf(ang);
        const float other = src[base + ((d < 32) ? d + 32 : d - 32)];
        v = (d < 32) ? (v * c - other * sn) : (v * c + other * sn);
    }
    *out = (bf16)v;
}

// V: [S, NKV*64] fp32 -> V^T [NKV][64][s_pad] bf16 (zero pad).
__global__ void transpose_vt_kernel(const float* __restrict__ src,
                                    bf16* __restrict__ dst, int s_pad)
{
    const int s = blockIdx.x;
    const int h = blockIdx.y;
    const int d = threadIdx.x;
    bf16* out = dst + ((size_t)h * HD + d) * s_pad + s;
    *out = (s >= S_LEN) ? (bf16)0.0f
                        : (bf16)src[(size_t)s * NKV * HD + (size_t)h * HD + d];
}

// KB K/V: prepend sink token, pad to KB_PAD, optional transpose (for V^T).
__global__ void kb_build_kernel(const float* __restrict__ proj,   // [NKB, NKV*64]
                                const float* __restrict__ sink,   // [NKV*64]
                                bf16* __restrict__ dst, int transpose)
{
    const int j = blockIdx.x;            // 0..KB_PAD-1
    const int h = blockIdx.y;
    const int d = threadIdx.x;
    float v = 0.0f;
    if (j == 0)          v = sink[h * HD + d];
    else if (j <= NKB)   v = proj[(size_t)(j - 1) * (NKV * HD) + (size_t)h * HD + d];
    const size_t idx = transpose ? ((size_t)h * HD + d) * KB_PAD + j
                                 : ((size_t)h * KB_PAD + j) * HD + d;
    dst[idx] = (bf16)v;
}

// ---------------------------------------------------------------------------
// Flash attention: one wave per (head, 16-row q tile).  Online softmax over
// [KB(257) ; causal self(<=qr0+16)] in 32-key chunks, all matmuls via WMMA.
// ---------------------------------------------------------------------------
struct AttnState {
    v8f o0, o1, o2, o3;
    float m[8], l[8];
};

__device__ __forceinline__ void attn_chunk(
    AttnState& st, v16bf aq0, v16bf aq1,
    const bf16* __restrict__ Kb,          // [*, 64] keys row-major (K-contig d)
    const bf16* __restrict__ Vt, int ldv, // [64, ldv] V^T (K-contig keys)
    int c, int limit, int rowBase, bool causal, int lane, bf16* pbuf)
{
    const int cl = lane & 15;
    const int hg = (lane >> 4) * 8;

    // scores: two 16x16 D tiles covering keys c..c+31 (K = HD = 64 -> 2 steps)
    v8f s0 = {}, s1 = {};
    s0 = wmma_bf16(aq0, load_frag(Kb, HD, c,       0, lane), s0);
    s0 = wmma_bf16(aq1, load_frag(Kb, HD, c,      32, lane), s0);
    s1 = wmma_bf16(aq0, load_frag(Kb, HD, c + 16,  0, lane), s1);
    s1 = wmma_bf16(aq1, load_frag(Kb, HD, c + 16, 32, lane), s1);

    const float scale = 0.125f;          // 1/sqrt(64)
    const int key0 = c + cl, key1 = c + 16 + cl;

    float x0[8], x1[8];
#pragma unroll
    for (int i = 0; i < 8; ++i) {
        const int row = rowBase + hg + i;
        float a = s0[i] * scale;
        float b = s1[i] * scale;
        if (key0 >= limit || (causal && key0 > row)) a = -1e30f;
        if (key1 >= limit || (causal && key1 > row)) b = -1e30f;
        x0[i] = a; x1[i] = b;
    }

#pragma unroll
    for (int i = 0; i < 8; ++i) {
        // row-wise max across the 16 lanes holding this row's columns
        float t = fmaxf(x0[i], x1[i]);
        t = fmaxf(t, __shfl_xor(t, 1, 32));
        t = fmaxf(t, __shfl_xor(t, 2, 32));
        t = fmaxf(t, __shfl_xor(t, 4, 32));
        t = fmaxf(t, __shfl_xor(t, 8, 32));
        const float mn   = fmaxf(st.m[i], t);
        const float corr = __expf(st.m[i] - mn);
        st.m[i] = mn;
        const float e0 = __expf(x0[i] - mn);
        const float e1 = __expf(x1[i] - mn);
        float rs = e0 + e1;
        rs += __shfl_xor(rs, 1, 32);
        rs += __shfl_xor(rs, 2, 32);
        rs += __shfl_xor(rs, 4, 32);
        rs += __shfl_xor(rs, 8, 32);
        st.l[i] = st.l[i] * corr + rs;
        st.o0[i] *= corr; st.o1[i] *= corr; st.o2[i] *= corr; st.o3[i] *= corr;
        // stage P (bf16) in LDS in [row][col] order for A-operand re-layout
        pbuf[(hg + i) * 32 + cl]      = (bf16)e0;
        pbuf[(hg + i) * 32 + 16 + cl] = (bf16)e1;
    }

    asm volatile("s_wait_dscnt 0" ::: "memory");

    // P as WMMA A operand: lane = row, K groups {hg..hg+7, 16+hg..16+hg+7}
    FragBF pa;
    const bf16* pp = pbuf + cl * 32 + hg;
    pa.h[0] = *reinterpret_cast<const v8bf*>(pp);
    pa.h[1] = *reinterpret_cast<const v8bf*>(pp + 16);

    // P[16x32] x V[32x64]: B operand from V^T, keys K-contiguous
    st.o0 = wmma_bf16(pa.v, load_frag(Vt, ldv,  0, c, lane), st.o0);
    st.o1 = wmma_bf16(pa.v, load_frag(Vt, ldv, 16, c, lane), st.o1);
    st.o2 = wmma_bf16(pa.v, load_frag(Vt, ldv, 32, c, lane), st.o2);
    st.o3 = wmma_bf16(pa.v, load_frag(Vt, ldv, 48, c, lane), st.o3);
}

__global__ __launch_bounds__(32) void attention_kernel(
    const bf16* __restrict__ qt,    // [NH][S][64]
    const bf16* __restrict__ kt,    // [NKV][S_PAD][64]
    const bf16* __restrict__ vt,    // [NKV][64][S_PAD]
    const bf16* __restrict__ kbqt,  // [NH][S][64]
    const bf16* __restrict__ kbkt,  // [NKV][KB_PAD][64]
    const bf16* __restrict__ kbvt,  // [NKV][64][KB_PAD]
    bf16* __restrict__ o_in)        // [S][NH*64]
{
    __shared__ __align__(16) bf16 pbuf[16 * 32];
    const int lane = threadIdx.x;
    const int h    = blockIdx.y;
    const int qr0  = blockIdx.x * 16;
    const int kvh  = h / GROUPS;

    AttnState st;
    st.o0 = v8f{}; st.o1 = v8f{}; st.o2 = v8f{}; st.o3 = v8f{};
#pragma unroll
    for (int i = 0; i < 8; ++i) { st.m[i] = -1e30f; st.l[i] = 0.0f; }

    // Phase 1: knowledge-base keys (no causal mask)
    {
        const bf16* aqb = kbqt + (size_t)h * S_LEN * HD;
        v16bf aq0 = load_frag(aqb, HD, qr0,  0, lane);
        v16bf aq1 = load_frag(aqb, HD, qr0, 32, lane);
        const bf16* Kb = kbkt + (size_t)kvh * KB_PAD * HD;
        const bf16* Vt = kbvt + (size_t)kvh * HD * KB_PAD;
        for (int c = 0; c < KB_LEN; c += 32)
            attn_chunk(st, aq0, aq1, Kb, Vt, KB_PAD, c, KB_LEN, 0, false, lane, pbuf);
    }
    // Phase 2: causal self attention
    {
        const bf16* aqb = qt + (size_t)h * S_LEN * HD;
        v16bf aq0 = load_frag(aqb, HD, qr0,  0, lane);
        v16bf aq1 = load_frag(aqb, HD, qr0, 32, lane);
        const bf16* Kb = kt + (size_t)kvh * S_PAD * HD;
        const bf16* Vt = vt + (size_t)kvh * HD * S_PAD;
        for (int c = 0; c < qr0 + 16; c += 32)
            attn_chunk(st, aq0, aq1, Kb, Vt, S_PAD, c, S_LEN, qr0, true, lane, pbuf);
    }

    const int cl = lane & 15;
    const int hg = (lane >> 4) * 8;
#pragma unroll
    for (int i = 0; i < 8; ++i) {
        const float inv = 1.0f / st.l[i];
        const size_t row = (size_t)(qr0 + hg + i);
        bf16* dst = o_in + row * (NH * HD) + (size_t)h * HD + cl;
        dst[ 0] = (bf16)(st.o0[i] * inv);
        dst[16] = (bf16)(st.o1[i] * inv);
        dst[32] = (bf16)(st.o2[i] * inv);
        dst[48] = (bf16)(st.o3[i] * inv);
    }
}

// ---------------------------------------------------------------------------
// Host launcher
// ---------------------------------------------------------------------------
static inline dim3 gemm_grid(int M, int N) {
    return dim3((unsigned)((M + 127) / 128), (unsigned)(N / 64));
}

extern "C" void kernel_launch(void* const* d_in, const int* in_sizes, int n_in,
                              void* d_out, int out_size, void* d_ws, size_t ws_size,
                              hipStream_t stream)
{
    (void)in_sizes; (void)n_in; (void)out_size; (void)ws_size;

    const float* hidden   = (const float*)d_in[0];
    const float* kb_keys  = (const float*)d_in[1];
    const float* kb_vals  = (const float*)d_in[2];
    const float* q_w      = (const float*)d_in[3];
    const float* k_w      = (const float*)d_in[4];
    const float* v_w      = (const float*)d_in[5];
    const float* o_w      = (const float*)d_in[6];
    const float* qnew_w   = (const float*)d_in[7];
    const float* kbk_w    = (const float*)d_in[8];
    const float* kbv_w    = (const float*)d_in[9];
    const float* sink_k   = (const float*)d_in[10];
    const float* sink_v   = (const float*)d_in[11];
    const float* ln1_w    = (const float*)d_in[12];
    const float* ln2_w    = (const float*)d_in[13];
    const float* up_w     = (const float*)d_in[14];
    const float* down_w   = (const float*)d_in[15];
    const int*   pos_ids  = (const int*)d_in[16];
    float* out = (float*)d_out;

    char* ws = (char*)d_ws;
    size_t off = 0;
    auto alloc = [&](size_t bytes) -> void* {
        void* p = ws + off;
        off += (bytes + 255) & ~(size_t)255;
        return p;
    };

    // bf16 weights
    bf16* wq   = (bf16*)alloc((size_t)H_DIM * H_DIM * 2);
    bf16* wk   = (bf16*)alloc((size_t)NKV * HD * H_DIM * 2);
    bf16* wv   = (bf16*)alloc((size_t)NKV * HD * H_DIM * 2);
    bf16* wo   = (bf16*)alloc((size_t)H_DIM * H_DIM * 2);
    bf16* wqn  = (bf16*)alloc((size_t)H_DIM * H_DIM * 2);
    bf16* wkbk = (bf16*)alloc((size_t)NKV * HD * H_DIM * 2);
    bf16* wkbv = (bf16*)alloc((size_t)NKV * HD * H_DIM * 2);
    bf16* wup  = (bf16*)alloc((size_t)DFF * H_DIM * 2);
    bf16* wdn  = (bf16*)alloc((size_t)H_DIM * DFF * 2);
    // bf16 activations / intermediates
    bf16* xbf    = (bf16*)alloc((size_t)S_LEN * H_DIM * 2);
    bf16* kbkbf  = (bf16*)alloc((size_t)NKB * H_DIM * 2);
    bf16* kbvbf  = (bf16*)alloc((size_t)NKB * H_DIM * 2);
    float* qf    = (float*)alloc((size_t)S_LEN * H_DIM * 4);
    float* kf    = (float*)alloc((size_t)S_LEN * NKV * HD * 4);
    float* vf    = (float*)alloc((size_t)S_LEN * NKV * HD * 4);
    float* kbqf  = (float*)alloc((size_t)S_LEN * H_DIM * 4);
    float* kbkf  = (float*)alloc((size_t)NKB * NKV * HD * 4);
    float* kbvf  = (float*)alloc((size_t)NKB * NKV * HD * 4);
    bf16* qt     = (bf16*)alloc((size_t)NH * S_LEN * HD * 2);
    bf16* kt     = (bf16*)alloc((size_t)NKV * S_PAD * HD * 2);
    bf16* vt     = (bf16*)alloc((size_t)NKV * HD * S_PAD * 2);
    bf16* kbqt   = (bf16*)alloc((size_t)NH * S_LEN * HD * 2);
    bf16* kbkt   = (bf16*)alloc((size_t)NKV * KB_PAD * HD * 2);
    bf16* kbvt   = (bf16*)alloc((size_t)NKV * HD * KB_PAD * 2);
    bf16* o_in   = (bf16*)alloc((size_t)S_LEN * H_DIM * 2);
    float* hbuf  = (float*)alloc((size_t)S_LEN * H_DIM * 4);
    bf16* ybf    = (bf16*)alloc((size_t)S_LEN * H_DIM * 2);
    float* ubuf  = (float*)alloc((size_t)S_LEN * DFF * 4);
    bf16* actbf  = (bf16*)alloc((size_t)S_LEN * DFF * 2);

    auto cvt = [&](const float* s, bf16* d, size_t n) {
        cvt_bf16_kernel<<<dim3((unsigned)((n + 255) / 256)), dim3(256), 0, stream>>>(s, d, n);
    };

    // 1) weight + KB-input conversion to bf16
    cvt(q_w,    wq,   (size_t)H_DIM * H_DIM);
    cvt(k_w,    wk,   (size_t)NKV * HD * H_DIM);
    cvt(v_w,    wv,   (size_t)NKV * HD * H_DIM);
    cvt(o_w,    wo,   (size_t)H_DIM * H_DIM);
    cvt(qnew_w, wqn,  (size_t)H_DIM * H_DIM);
    cvt(kbk_w,  wkbk, (size_t)NKV * HD * H_DIM);
    cvt(kbv_w,  wkbv, (size_t)NKV * HD * H_DIM);
    cvt(up_w,   wup,  (size_t)DFF * H_DIM);
    cvt(down_w, wdn,  (size_t)H_DIM * DFF);
    cvt(kb_keys, kbkbf, (size_t)NKB * H_DIM);
    cvt(kb_vals, kbvbf, (size_t)NKB * H_DIM);

    // 2) RMSNorm 1
    rmsnorm_kernel<<<dim3(S_LEN), dim3(256), 0, stream>>>(hidden, ln1_w, xbf);

    // 3) projections (C = A * W^T)
    gemm_bf16_nt<<<gemm_grid(S_LEN, H_DIM),     dim3(256), 0, stream>>>(xbf, wq,  qf,   S_LEN, H_DIM,    H_DIM, nullptr);
    gemm_bf16_nt<<<gemm_grid(S_LEN, NKV * HD),  dim3(256), 0, stream>>>(xbf, wk,  kf,   S_LEN, NKV * HD, H_DIM, nullptr);
    gemm_bf16_nt<<<gemm_grid(S_LEN, NKV * HD),  dim3(256), 0, stream>>>(xbf, wv,  vf,   S_LEN, NKV * HD, H_DIM, nullptr);
    gemm_bf16_nt<<<gemm_grid(S_LEN, H_DIM),     dim3(256), 0, stream>>>(xbf, wqn, kbqf, S_LEN, H_DIM,    H_DIM, nullptr);
    gemm_bf16_nt<<<gemm_grid(NKB, NKV * HD),    dim3(256), 0, stream>>>(kbkbf, wkbk, kbkf, NKB, NKV * HD, H_DIM, nullptr);
    gemm_bf16_nt<<<gemm_grid(NKB, NKV * HD),    dim3(256), 0, stream>>>(kbvbf, wkbv, kbvf, NKB, NKV * HD, H_DIM, nullptr);

    // 4) RoPE + layout transforms
    rope_transpose_kernel<<<dim3(S_LEN, NH),   dim3(HD), 0, stream>>>(qf,   qt,   pos_ids, NH,  S_LEN, 1);
    rope_transpose_kernel<<<dim3(S_PAD, NKV),  dim3(HD), 0, stream>>>(kf,   kt,   pos_ids, NKV, S_PAD, 1);
    transpose_vt_kernel  <<<dim3(S_PAD, NKV),  dim3(HD), 0, stream>>>(vf,   vt,   S_PAD);
    rope_transpose_kernel<<<dim3(S_LEN, NH),   dim3(HD), 0, stream>>>(kbqf, kbqt, pos_ids, NH,  S_LEN, 0);
    kb_build_kernel      <<<dim3(KB_PAD, NKV), dim3(HD), 0, stream>>>(kbkf, sink_k, kbkt, 0);
    kb_build_kernel      <<<dim3(KB_PAD, NKV), dim3(HD), 0, stream>>>(kbvf, sink_v, kbvt, 1);

    // 5) joint rectangular flash attention
    attention_kernel<<<dim3(S_LEN / 16, NH), dim3(32), 0, stream>>>(
        qt, kt, vt, kbqt, kbkt, kbvt, o_in);

    // 6) output projection + residual -> h
    gemm_bf16_nt<<<gemm_grid(S_LEN, H_DIM), dim3(256), 0, stream>>>(
        o_in, wo, hbuf, S_LEN, H_DIM, H_DIM, hidden);

    // 7) MLP: rmsnorm -> up -> relu^2 -> down (+h) -> out
    rmsnorm_kernel<<<dim3(S_LEN), dim3(256), 0, stream>>>(hbuf, ln2_w, ybf);
    gemm_bf16_nt<<<gemm_grid(S_LEN, DFF), dim3(256), 0, stream>>>(
        ybf, wup, ubuf, S_LEN, DFF, H_DIM, nullptr);
    sqrelu_bf16_kernel<<<dim3((unsigned)(((size_t)S_LEN * DFF + 255) / 256)), dim3(256), 0, stream>>>(
        ubuf, actbf, (size_t)S_LEN * DFF);
    gemm_bf16_nt<<<gemm_grid(S_LEN, H_DIM), dim3(256), 0, stream>>>(
        actbf, wdn, out, S_LEN, H_DIM, DFF, hbuf);
}
